// MultiHeadSelfAttention_40638980555387
// MI455X (gfx1250) — matrix operand order repaired and checked by reading, not compiled, not run on hardware
//
#include <hip/hip_runtime.h>
#include <hip/hip_bf16.h>

#define N_TOK   4096
#define D_EMB   768
#define H_NUM   12
#define D_HEAD  64
#define D_QKV   2304
#define ATT_SCALE 0.125f   // 64^-0.5

typedef __attribute__((ext_vector_type(16))) __bf16 v16bf;
typedef __attribute__((ext_vector_type(8)))  __bf16 v8bf;
typedef __attribute__((ext_vector_type(8)))  float  v8f;
typedef __attribute__((ext_vector_type(4)))  unsigned int v4u32;
typedef __attribute__((ext_vector_type(8)))  int v8i32;
typedef __attribute__((ext_vector_type(4)))  int v4i32;

static __device__ __forceinline__ v16bf join8(v8bf lo, v8bf hi) {
  return __builtin_shufflevector(lo, hi, 0,1,2,3,4,5,6,7,8,9,10,11,12,13,14,15);
}

// LDS offset (bytes, workgroup-relative) of a generic pointer to __shared__.
static __device__ __forceinline__ unsigned lds_off(const void* p) {
  return (unsigned)(unsigned long long)p;   // generic LDS addr[31:0] == LDS offset
}

// Async global->LDS copy of 16 bytes per lane (ASYNCcnt path). The 24-bit
// instruction offset is added to BOTH the global and the LDS address.
static __device__ __forceinline__ void async_b128(unsigned lds, const void* g) {
  asm volatile("global_load_async_to_lds_b128 %0, %1, off" :: "v"(lds), "v"(g) : "memory");
}
static __device__ __forceinline__ void async_b128_o16(unsigned lds, const void* g) {
  asm volatile("global_load_async_to_lds_b128 %0, %1, off offset:16" :: "v"(lds), "v"(g) : "memory");
}
static __device__ __forceinline__ void wait_async(int n) {
  if (n == 0) asm volatile("s_wait_asynccnt 0" ::: "memory");
  else        asm volatile("s_wait_asynccnt 4" ::: "memory");
}

// ---------------------------------------------------------------------------
// Tensor Data Mover: 2-D bf16 tile (tile_x contiguous elems  x  tile_y lines)
// from a row-major tensor with pitch stride_x (elements) into LDS at lds.
// D# layout per CDNA5 ISA ch.8; 6-arg builtin (g0, g1, g2, g3, g4, cpol).
// ---------------------------------------------------------------------------
static __device__ __forceinline__ void tdm_load_tile_2d_bf16(
    unsigned lds, const __bf16* gsrc,
    unsigned tile_x, unsigned tile_y,
    unsigned tensor_x, unsigned tensor_y,
    unsigned long long stride_x)
{
  unsigned long long ga = (unsigned long long)(const void*)gsrc;
  v4u32 g0;
  g0.x = 1u;                                              // count=1, user D#
  g0.y = lds;                                             // lds_addr
  g0.z = (unsigned)ga;                                    // global_addr[31:0]
  g0.w = (unsigned)((ga >> 32) & 0x01FFFFFFu) | (2u << 30); // addr[56:32], type=2
  v8i32 g1;
  g1.s0 = (int)(1u << 16);                                // data_size=1 -> 2 bytes
  g1.s1 = (int)((tensor_x & 0xFFFFu) << 16);              // tensor_dim0[15:0]
  g1.s2 = (int)((tensor_x >> 16) | ((tensor_y & 0xFFFFu) << 16));
  g1.s3 = (int)((tensor_y >> 16) | (tile_x << 16));       // tile_dim0
  g1.s4 = (int)tile_y;                                    // tile_dim1 (tile_dim2=0)
  g1.s5 = (int)(unsigned)stride_x;                        // tensor_dim0_stride lo
  g1.s6 = (int)(unsigned)(stride_x >> 32);                // stride hi, dim1_stride=0
  g1.s7 = 0;
  v4i32 gz4 = {0, 0, 0, 0};
  v8i32 gz8 = {0, 0, 0, 0, 0, 0, 0, 0};
  __builtin_amdgcn_tensor_load_to_lds(g0, g1, gz4, gz4, gz8, 0);
}

// ---------------------------------------------------------------------------
// f32 -> bf16 conversion (4 elems / thread, exact grid)
// ---------------------------------------------------------------------------
__global__ void cvt_f32_bf16(const float* __restrict__ src,
                             __bf16* __restrict__ dst, int n) {
  int i = (blockIdx.x * blockDim.x + threadIdx.x) * 4;
  if (i + 3 < n) {
    float4 f = *(const float4*)(src + i);
    dst[i + 0] = (__bf16)f.x;
    dst[i + 1] = (__bf16)f.y;
    dst[i + 2] = (__bf16)f.z;
    dst[i + 3] = (__bf16)f.w;
  }
}

// ---------------------------------------------------------------------------
// 64x64 block-tile bf16 GEMM mainloop: C[64,64] = A[64,K] * B[64,K]^T.
// Double-buffered ASYNC global->LDS staging: next K-slice's loads stay in
// flight (s_wait_asynccnt 4) while the current slice feeds the WMMAs.
// 4 waves, each computes a 32x32 sub-tile as 2x2 WMMA 16x16 tiles.
// ---------------------------------------------------------------------------
__device__ __forceinline__ void gemm_tile_64x64(
    const __bf16* __restrict__ A, const __bf16* __restrict__ B,
    int K, int bM, int bN,
    __bf16* As0, __bf16* As1, __bf16* Bs0, __bf16* Bs1, v8f c[2][2])
{
  const int tid  = threadIdx.x;
  const int lane = tid & 31;
  const int wid  = tid >> 5;
  const int wm   = (wid >> 1) * 32;
  const int wn   = (wid & 1) * 32;
  const int ln   = lane & 15;
  const int c0   = (lane < 16) ? 0 : 16;   // A-frag K-chunk base
  const int kb0  = (lane < 16) ? 0 : 16;   // B-frag K base
  const int lrow = tid >> 1;               // 0..63 (fill row)
  const int lseg = (tid & 1) * 16;         // 0 / 16 (fill K segment)

  const unsigned asL[2] = { lds_off(As0 + lrow * 32 + lseg),
                            lds_off(As1 + lrow * 32 + lseg) };
  const unsigned bsL[2] = { lds_off(Bs0 + lrow * 32 + lseg),
                            lds_off(Bs1 + lrow * 32 + lseg) };
  const __bf16* ag = A + (bM + lrow) * K + lseg;
  const __bf16* bg = B + (bN + lrow) * K + lseg;
  const int KT = K >> 5;

  // prologue: stage slice 0 into buffer 0
  async_b128    (asL[0], ag);
  async_b128_o16(asL[0], ag);
  async_b128    (bsL[0], bg);
  async_b128_o16(bsL[0], bg);

  for (int kt = 0; kt < KT; ++kt) {
    const int cur = kt & 1;
    if (kt + 1 < KT) {                     // prefetch next slice into other buf
      const int nxt = cur ^ 1;
      const __bf16* a2 = ag + (kt + 1) * 32;
      const __bf16* b2 = bg + (kt + 1) * 32;
      async_b128    (asL[nxt], a2);
      async_b128_o16(asL[nxt], a2);
      async_b128    (bsL[nxt], b2);
      async_b128_o16(bsL[nxt], b2);
      wait_async(4);                       // current slice done, next in flight
    } else {
      wait_async(0);
    }
    __syncthreads();

    const __bf16* Asb = cur ? As1 : As0;
    const __bf16* Bsb = cur ? Bs1 : Bs0;

    v16bf a[2], b[2];
#pragma unroll
    for (int i = 0; i < 2; ++i) {
      int r = wm + i * 16 + ln;
      a[i] = join8(*(const v8bf*)&Asb[r * 32 + c0],
                   *(const v8bf*)&Asb[r * 32 + c0 + 8]);
    }
#pragma unroll
    for (int j = 0; j < 2; ++j) {
      int r = wn + j * 16 + ln;
      b[j] = *(const v16bf*)&Bsb[r * 32 + kb0];
    }
#pragma unroll
    for (int i = 0; i < 2; ++i)
#pragma unroll
      for (int j = 0; j < 2; ++j)
        c[i][j] = __builtin_amdgcn_wmma_f32_16x16x32_bf16(
            false, a[i], false, b[j], (short)0, c[i][j], false, false);

    __syncthreads();                       // buf reads done before re-fill
  }
}

// ---------------------------------------------------------------------------
// QKV projection: [4096,768] x [2304,768]^T + b  -> per-head Q, K, V^T (bf16)
// ---------------------------------------------------------------------------
__global__ void qkv_gemm(const __bf16* __restrict__ xb,
                         const __bf16* __restrict__ wb,
                         const float*  __restrict__ bias,
                         __bf16* __restrict__ qb,
                         __bf16* __restrict__ kb,
                         __bf16* __restrict__ vtb)
{
  __shared__ alignas(128) __bf16 As[2][64 * 32];
  __shared__ alignas(128) __bf16 Bs[2][64 * 32];
  v8f c[2][2] = {};

  const int bM = blockIdx.x * 64;
  const int bN = blockIdx.y * 64;
  gemm_tile_64x64(xb, wb, D_EMB, bM, bN, As[0], As[1], Bs[0], Bs[1], c);

  const int lane = threadIdx.x & 31;
  const int wid  = threadIdx.x >> 5;
  const int wm   = (wid >> 1) * 32;
  const int wn   = (wid & 1) * 32;
  const int ln   = lane & 15;
  const int half = lane >> 4;

#pragma unroll
  for (int i = 0; i < 2; ++i) {
#pragma unroll
    for (int j = 0; j < 2; ++j) {
      const int col   = bN + wn + j * 16 + ln;          // 0..2303
      const float bv  = bias[col];
      const int which = col / D_EMB;                    // 0=q 1=k 2=v
      const int d     = col - which * D_EMB;
      const int h     = d >> 6;
      const int dh    = d & 63;
#pragma unroll
      for (int r = 0; r < 8; ++r) {
        const int row = bM + wm + i * 16 + r + half * 8;
        const float v = c[i][j][r] + bv;
        if (which == 0)      qb[(h * N_TOK + row) * D_HEAD + dh] = (__bf16)v;
        else if (which == 1) kb[(h * N_TOK + row) * D_HEAD + dh] = (__bf16)v;
        else                 vtb[(h * D_HEAD + dh) * N_TOK + row] = (__bf16)v;
      }
    }
  }
}

// ---------------------------------------------------------------------------
// Flash attention: grid (N/64, H). 4 waves, each owns 16 query rows.
// K / V^T tiles streamed by the Tensor Data Mover, double-buffered: the next
// key block's two descriptors stay in flight (s_wait_tensorcnt 2) while the
// current block runs S = QK^T, online softmax and O += PV.
// ---------------------------------------------------------------------------
__global__ void attn_kernel(const __bf16* __restrict__ qb,
                            const __bf16* __restrict__ kb,
                            const __bf16* __restrict__ vtb,
                            __bf16* __restrict__ ctxb)
{
  __shared__ alignas(128) __bf16 Ks[2][64 * 64];    // [key][dh]
  __shared__ alignas(128) __bf16 Vs[2][64 * 64];    // [dh][key]  (V^T)
  __shared__ alignas(128) __bf16 Ps[4][16 * 64];    // per-wave P tile

  const int h    = blockIdx.y;
  const int qblk = blockIdx.x;
  const int tid  = threadIdx.x;
  const int lane = tid & 31;
  const int wid  = tid >> 5;
  const int ln   = lane & 15;
  const int half = lane >> 4;
  const int c0   = half * 16;
  const int kb0  = half * 16;

  const unsigned ksL[2] = { lds_off(Ks[0]), lds_off(Ks[1]) };
  const unsigned vsL[2] = { lds_off(Vs[0]), lds_off(Vs[1]) };
  const __bf16* kg = kb  + h * N_TOK * D_HEAD;      // K head base  [4096 x 64]
  const __bf16* vg = vtb + h * D_HEAD * N_TOK;      // V^T head base [64 x 4096]

  // Q fragments for this wave's 16 rows (reused across all key blocks)
  const __bf16* qrowp = qb + (h * N_TOK + qblk * 64 + wid * 16 + ln) * D_HEAD;
  const v16bf qf0 = join8(*(const v8bf*)(qrowp + c0),
                          *(const v8bf*)(qrowp + c0 + 8));
  const v16bf qf1 = join8(*(const v8bf*)(qrowp + 32 + c0),
                          *(const v8bf*)(qrowp + 32 + c0 + 8));

  v8f o[4] = {};
  float m[8], l[8];
#pragma unroll
  for (int r = 0; r < 8; ++r) { m[r] = -1e30f; l[r] = 0.f; }

  const int NBLK = N_TOK / 64;
  if (wid == 0) {      // prologue: stage key block 0 into buffer 0
    tdm_load_tile_2d_bf16(ksL[0], kg, 64, 64, D_HEAD, N_TOK, D_HEAD);
    tdm_load_tile_2d_bf16(vsL[0], vg, 64, 64, N_TOK, D_HEAD, N_TOK);
  }

  for (int j = 0; j < NBLK; ++j) {
    const int cur = j & 1;
    if (wid == 0) {
      if (j + 1 < NBLK) {                  // prefetch next key block
        const int nxt = cur ^ 1;
        tdm_load_tile_2d_bf16(ksL[nxt], kg + (j + 1) * 64 * D_HEAD,
                              64, 64, D_HEAD, N_TOK, D_HEAD);
        tdm_load_tile_2d_bf16(vsL[nxt], vg + (j + 1) * 64,
                              64, 64, N_TOK, D_HEAD, N_TOK);
        __builtin_amdgcn_s_wait_tensorcnt(2);  // cur done, next in flight
      } else {
        __builtin_amdgcn_s_wait_tensorcnt(0);
      }
    }
    __syncthreads();

    const __bf16* Kst = Ks[cur];
    const __bf16* Vst = Vs[cur];

    // S = Q K^T   (16 rows x 64 keys, 4 WMMA tiles, 2 K-steps each)
    v8f s[4];
#pragma unroll
    for (int ct = 0; ct < 4; ++ct) {
      const int key = ct * 16 + ln;
      const v16bf b0 = *(const v16bf*)&Kst[key * 64 + kb0];
      const v16bf b1 = *(const v16bf*)&Kst[key * 64 + kb0 + 32];
      v8f z = {};
      z = __builtin_amdgcn_wmma_f32_16x16x32_bf16(false, qf0, false, b0, (short)0, z, false, false);
      z = __builtin_amdgcn_wmma_f32_16x16x32_bf16(false, qf1, false, b1, (short)0, z, false, false);
      s[ct] = z;
    }

    // online softmax: row max / row sum within 16-lane halves
    float mnew[8], fac[8];
#pragma unroll
    for (int r = 0; r < 8; ++r) {
      float t = fmaxf(fmaxf(s[0][r], s[1][r]), fmaxf(s[2][r], s[3][r]));
#pragma unroll
      for (int x = 1; x < 16; x <<= 1) t = fmaxf(t, __shfl_xor(t, x, 32));
      mnew[r] = fmaxf(m[r], t * ATT_SCALE);
      fac[r]  = __expf(m[r] - mnew[r]);
    }
    float rs[8];
#pragma unroll
    for (int r = 0; r < 8; ++r) rs[r] = 0.f;
#pragma unroll
    for (int ct = 0; ct < 4; ++ct) {
#pragma unroll
      for (int r = 0; r < 8; ++r) {
        const float p = __expf(s[ct][r] * ATT_SCALE - mnew[r]);
        rs[r] += p;
        Ps[wid][(r + half * 8) * 64 + ct * 16 + ln] = (__bf16)p;
      }
    }
#pragma unroll
    for (int r = 0; r < 8; ++r) {
      float t = rs[r];
#pragma unroll
      for (int x = 1; x < 16; x <<= 1) t += __shfl_xor(t, x, 32);
      l[r] = l[r] * fac[r] + t;
      m[r] = mnew[r];
    }
#pragma unroll
    for (int n = 0; n < 4; ++n)
#pragma unroll
      for (int r = 0; r < 8; ++r) o[n][r] *= fac[r];

    __syncthreads();   // P tiles visible for A-layout re-read

    // O += P V  (P re-read as A fragments, V^T rows as B fragments)
    const __bf16* pw = &Ps[wid][0];
    const v16bf p0 = join8(*(const v8bf*)(pw + ln * 64 + c0),
                           *(const v8bf*)(pw + ln * 64 + c0 + 8));
    const v16bf p1 = join8(*(const v8bf*)(pw + ln * 64 + 32 + c0),
                           *(const v8bf*)(pw + ln * 64 + 32 + c0 + 8));
#pragma unroll
    for (int n = 0; n < 4; ++n) {
      const int dh = n * 16 + ln;
      const v16bf v0 = *(const v16bf*)&Vst[dh * 64 + kb0];
      const v16bf v1 = *(const v16bf*)&Vst[dh * 64 + kb0 + 32];
      o[n] = __builtin_amdgcn_wmma_f32_16x16x32_bf16(false, p0, false, v0, (short)0, o[n], false, false);
      o[n] = __builtin_amdgcn_wmma_f32_16x16x32_bf16(false, p1, false, v1, (short)0, o[n], false, false);
    }

    __syncthreads();   // buf reads done before TDM re-fills it next iter
  }

  // normalize and write ctx (bf16, [N, D] with col = h*64+dh)
#pragma unroll
  for (int n = 0; n < 4; ++n) {
    const int col = h * D_HEAD + n * 16 + ln;
#pragma unroll
    for (int r = 0; r < 8; ++r) {
      const int row = qblk * 64 + wid * 16 + r + half * 8;
      ctxb[row * D_EMB + col] = (__bf16)(o[n][r] / l[r]);
    }
  }
}

// ---------------------------------------------------------------------------
// Output projection: [4096,768] x [768,768]^T + b -> f32 out
// ---------------------------------------------------------------------------
__global__ void out_gemm(const __bf16* __restrict__ ctxb,
                         const __bf16* __restrict__ wb,
                         const float*  __restrict__ bias,
                         float* __restrict__ out)
{
  __shared__ alignas(128) __bf16 As[2][64 * 32];
  __shared__ alignas(128) __bf16 Bs[2][64 * 32];
  v8f c[2][2] = {};

  const int bM = blockIdx.x * 64;
  const int bN = blockIdx.y * 64;
  gemm_tile_64x64(ctxb, wb, D_EMB, bM, bN, As[0], As[1], Bs[0], Bs[1], c);

  const int lane = threadIdx.x & 31;
  const int wid  = threadIdx.x >> 5;
  const int wm   = (wid >> 1) * 32;
  const int wn   = (wid & 1) * 32;
  const int ln   = lane & 15;
  const int half = lane >> 4;

#pragma unroll
  for (int i = 0; i < 2; ++i)
#pragma unroll
    for (int j = 0; j < 2; ++j) {
      const int col  = bN + wn + j * 16 + ln;
      const float bv = bias[col];
#pragma unroll
      for (int r = 0; r < 8; ++r) {
        const int row = bM + wm + i * 16 + r + half * 8;
        out[row * D_EMB + col] = c[i][j][r] + bv;
      }
    }
}

// ---------------------------------------------------------------------------
extern "C" void kernel_launch(void* const* d_in, const int* in_sizes, int n_in,
                              void* d_out, int out_size, void* d_ws, size_t ws_size,
                              hipStream_t stream) {
  const float* x    = (const float*)d_in[0];   // [1,4096,768]
  const float* wqkv = (const float*)d_in[1];   // [2304,768]
  const float* bqkv = (const float*)d_in[2];   // [2304]
  const float* wout = (const float*)d_in[3];   // [768,768]
  const float* bout = (const float*)d_in[4];   // [768]
  float* out = (float*)d_out;

  char* w = (char*)d_ws;
  __bf16* xb    = (__bf16*)(w);                     // 4096*768*2  = 6291456
  __bf16* wqkvb = (__bf16*)(w + 6291456);           // 2304*768*2  = 3538944
  __bf16* wob   = (__bf16*)(w + 9830400);           // 768*768*2   = 1179648
  __bf16* qb    = (__bf16*)(w + 11010048);          // 12*4096*64*2 = 6291456
  __bf16* kb    = (__bf16*)(w + 17301504);
  __bf16* vtb   = (__bf16*)(w + 23592960);
  __bf16* ctxb  = (__bf16*)(w + 29884416);          // end = 36175872

  cvt_f32_bf16<<<(N_TOK * D_EMB) / 1024, 256, 0, stream>>>(x, xb, N_TOK * D_EMB);
  cvt_f32_bf16<<<(D_QKV * D_EMB) / 1024, 256, 0, stream>>>(wqkv, wqkvb, D_QKV * D_EMB);
  cvt_f32_bf16<<<(D_EMB * D_EMB) / 1024, 256, 0, stream>>>(wout, wob, D_EMB * D_EMB);

  qkv_gemm<<<dim3(N_TOK / 64, D_QKV / 64), 128, 0, stream>>>(xb, wqkvb, bqkv, qb, kb, vtb);
  attn_kernel<<<dim3(N_TOK / 64, H_NUM), 128, 0, stream>>>(qb, kb, vtb, ctxb);
  out_gemm<<<dim3(N_TOK / 64, D_EMB / 64), 128, 0, stream>>>(ctxb, wob, bout, out);
}
